// DecoderRNN_10514079941046
// MI455X (gfx1250) — matrix operand order repaired and checked by reading, not compiled, not run on hardware
//
#include <hip/hip_runtime.h>
#include <hip/hip_bf16.h>
#include <math.h>

// Problem dimensions (fixed by the reference).
constexpr int kV = 50257;
constexpr int kE = 512;
constexpr int kH = 1024;
constexpr int kB = 64;
constexpr int kS = 512;

// ---------------------------------------------------------------------------
// WMMA vector types (CDNA5 / gfx1250, wave32)
// ---------------------------------------------------------------------------
typedef __attribute__((ext_vector_type(16))) __bf16          v16bf;
typedef __attribute__((ext_vector_type(16))) unsigned short  v16u;
typedef __attribute__((ext_vector_type(8)))  float           v8f;

// ---------------------------------------------------------------------------
// Split-bf16 helpers: x = hi + lo, each bf16 (round-to-nearest-even).
// Products Ahi*Bhi + Ahi*Blo + Alo*Bhi in f32 accumulators give ~2^-16
// relative accuracy: near-f32 GEMM at 3x bf16-WMMA cost.
// ---------------------------------------------------------------------------
__device__ __forceinline__ unsigned short f32_to_bf16_rne(float f) {
  unsigned int u = __builtin_bit_cast(unsigned int, f);
  unsigned int r = u + 0x7FFFu + ((u >> 16) & 1u);
  return (unsigned short)(r >> 16);
}
__device__ __forceinline__ float bf16_bits_to_f32(unsigned short h) {
  unsigned int u = ((unsigned int)h) << 16;
  return __builtin_bit_cast(float, u);
}
__device__ __forceinline__ void split_bf16(float x, unsigned short& hi, unsigned short& lo) {
  hi = f32_to_bf16_rne(x);
  lo = f32_to_bf16_rne(x - bf16_bits_to_f32(hi));
}

// 3-product accumulation (drops lo*lo term).
__device__ __forceinline__ v8f wmma3(v16bf ahi, v16bf alo, v16bf bhi, v16bf blo, v8f c) {
  c = __builtin_amdgcn_wmma_f32_16x16x32_bf16(false, ahi, false, bhi, (short)0, c, false, false);
  c = __builtin_amdgcn_wmma_f32_16x16x32_bf16(false, ahi, false, blo, (short)0, c, false, false);
  c = __builtin_amdgcn_wmma_f32_16x16x32_bf16(false, alo, false, bhi, (short)0, c, false, false);
  return c;
}

// ---------------------------------------------------------------------------
// Fragment layouts used here (contiguous k order, not the ISA table order; wave32):
//
// A (16x32 bf16, MxK): lane L: m = L%16; laneh = L/16
//   vector elem 2j,2j+1 -> K = 2j + 16*laneh (+0/+1)
//
// B (32x16 bf16, KxN): lane L: n = L%16; laneh = L/16
//   vector elem 2j,2j+1 -> K = 2j + 16*laneh (+0/+1)
// ---------------------------------------------------------------------------
__device__ __forceinline__ void load_a_frag_lds(const unsigned short (*Ahi)[128],
                                                const unsigned short (*Alo)[128],
                                                int m, int ks, int laneh,
                                                v16bf& hi, v16bf& lo) {
  v16u uh, ul;
#pragma unroll
  for (int j = 0; j < 8; ++j) {
    const int kk = ks + 2 * j + 16 * laneh;
    uh[2 * j]     = Ahi[m][kk];
    uh[2 * j + 1] = Ahi[m][kk + 1];
    ul[2 * j]     = Alo[m][kk];
    ul[2 * j + 1] = Alo[m][kk + 1];
  }
  hi = __builtin_bit_cast(v16bf, uh);
  lo = __builtin_bit_cast(v16bf, ul);
}

// On-the-fly B fragment from f32 global rows (used only for HBM-bound logits
// GEMM where pre-conversion would add traffic).
__device__ __forceinline__ void load_b_frag_global_f32(const float* __restrict__ rowptr,
                                                       v16bf& hi, v16bf& lo) {
  v16u uh, ul;
#pragma unroll
  for (int j = 0; j < 8; ++j) {
    const float2 f2 = *(const float2*)(rowptr + 2 * j);
    unsigned short h0, l0, h1, l1;
    split_bf16(f2.x, h0, l0);
    split_bf16(f2.y, h1, l1);
    uh[2 * j] = h0; uh[2 * j + 1] = h1;
    ul[2 * j] = l0; ul[2 * j + 1] = l1;
  }
  hi = __builtin_bit_cast(v16bf, uh);
  lo = __builtin_bit_cast(v16bf, ul);
}

// Stage a 64x128 f32 chunk into LDS as split hi/lo bf16. 256 threads.
__device__ __forceinline__ void stage_split64(const float* __restrict__ src,
                                              int k0,
                                              unsigned short (*Ahi)[128],
                                              unsigned short (*Alo)[128],
                                              int tid) {
#pragma unroll
  for (int off = 0; off < 64 * 128; off += 1024) {
    const int idx = off + tid * 4;
    const int m = idx >> 7;
    const int kk = idx & 127;
    const float4 f = *(const float4*)(src + (size_t)m * kH + k0 + kk);
    unsigned short h, l;
    split_bf16(f.x, h, l); Ahi[m][kk]     = h; Alo[m][kk]     = l;
    split_bf16(f.y, h, l); Ahi[m][kk + 1] = h; Alo[m][kk + 1] = l;
    split_bf16(f.z, h, l); Ahi[m][kk + 2] = h; Alo[m][kk + 2] = l;
    split_bf16(f.w, h, l); Ahi[m][kk + 3] = h; Alo[m][kk + 3] = l;
  }
}

// ---------------------------------------------------------------------------
// Kernel 0: pre-swizzle W (H x H) into WMMA B-fragment order, split bf16.
// B[k][n] = W[n][k].  Fragment (ntile, kstep) = 32 lanes x 16 bf16,
// stored contiguously (512 elements): frag idx = (ntile*32 + kstep).
// One thread builds one lane's 16 elements; writes are fully coalesced 32B.
// grid = (kH/16)*(kH/32)*32 / 256 = 256 blocks.
// ---------------------------------------------------------------------------
__global__ __launch_bounds__(256) void swizzle_w_kernel(const float* __restrict__ W,
                                                        unsigned short* __restrict__ Whi,
                                                        unsigned short* __restrict__ Wlo) {
  const int gid = blockIdx.x * 256 + threadIdx.x;  // (ntile*32 + kstep)*32 + lane
  const int lane = gid & 31;
  const int tile = gid >> 5;
  const int kstep = tile & 31;   // kH/32 = 32 k-steps
  const int ntile = tile >> 5;   // kH/16 = 64 n-tiles
  const int n = ntile * 16 + (lane & 15);
  const int kbase = kstep * 32 + 16 * (lane >> 4);
  const float* src = W + (size_t)n * kH + kbase;
  v16u hi, lo;
#pragma unroll
  for (int j = 0; j < 8; ++j) {
    const float2 f2 = *(const float2*)(src + 2 * j);
    unsigned short h0, l0, h1, l1;
    split_bf16(f2.x, h0, l0);
    split_bf16(f2.y, h1, l1);
    hi[2 * j] = h0; hi[2 * j + 1] = h1;
    lo[2 * j] = l0; lo[2 * j + 1] = l1;
  }
  *(v16u*)(Whi + (size_t)gid * 16) = hi;
  *(v16u*)(Wlo + (size_t)gid * 16) = lo;
}

// ---------------------------------------------------------------------------
// Kernel 1: Uh[b,i] = sum_j U[i,j] * hidden[0,b,j]
// ---------------------------------------------------------------------------
__global__ __launch_bounds__(256) void uh_kernel(const float* __restrict__ U,
                                                 const float* __restrict__ hidden,
                                                 float* __restrict__ Uh) {
  __shared__ float hl[kH];
  const int b = blockIdx.x >> 2;
  const int chunk = blockIdx.x & 3;
  for (int j = threadIdx.x; j < kH; j += 256) hl[j] = hidden[(size_t)b * kH + j];
  __syncthreads();
  const int i = chunk * 256 + threadIdx.x;
  const float* urow = U + (size_t)i * kH;
  float s = 0.f;
  for (int j = 0; j < kH; j += 4) {
    const float4 u4 = *(const float4*)(urow + j);
    s += u4.x * hl[j] + u4.y * hl[j + 1] + u4.z * hl[j + 2] + u4.w * hl[j + 3];
  }
  Uh[(size_t)b * kH + i] = s;
}

// ---------------------------------------------------------------------------
// Kernel 2: attention scores with fused tanh + v-dot epilogue.
//   e[b,s] = sum_h v[h] * tanh( (W @ enc[b,s])[h] + Uh[b,h] )
// GEMM: C(32768 x 1024) = enc(32768 x 1024) * W^T, fused N-reduction.
// B fragments come pre-swizzled (split bf16) -> two b128 loads, no VALU.
// grid = 512 WGs (64 rows each; rows never cross a batch), 256 threads.
// ---------------------------------------------------------------------------
__global__ __launch_bounds__(256) void attn_scores_kernel(const float* __restrict__ enc,
                                                          const unsigned short* __restrict__ Whi,
                                                          const unsigned short* __restrict__ Wlo,
                                                          const float* __restrict__ Uh,
                                                          const float* __restrict__ vvec,
                                                          float* __restrict__ e) {
  __shared__ unsigned short Ahi[64][128];
  __shared__ unsigned short Alo[64][128];
  __shared__ float e_acc[64][8];  // per-wave partial columns (deterministic)

  const int tid = threadIdx.x;
  const int wave = tid >> 5;
  const int lane = tid & 31;
  const int lane16 = lane & 15;
  const int laneh = lane >> 4;

  const int wg = blockIdx.x;
  const int b = wg >> 3;               // 8 WGs per batch (512/64)
  const int s_base = (wg & 7) * 64;
  const float* Arows = enc + ((size_t)b * kS + s_base) * kH;

  for (int i = tid; i < 64 * 8; i += 256) (&e_acc[0][0])[i] = 0.f;

  for (int npass = 0; npass < 2; ++npass) {
    const int n_wave_base = npass * 512 + wave * 64;
    const int ntile_base = n_wave_base >> 4;

    v8f C[4][4];
#pragma unroll
    for (int mt = 0; mt < 4; ++mt)
#pragma unroll
      for (int nt = 0; nt < 4; ++nt)
        C[mt][nt] = (v8f){0.f, 0.f, 0.f, 0.f, 0.f, 0.f, 0.f, 0.f};

    for (int k0 = 0; k0 < kH; k0 += 128) {
      __syncthreads();
      stage_split64(Arows, k0, Ahi, Alo, tid);
      __syncthreads();

#pragma unroll
      for (int ks = 0; ks < 128; ks += 32) {
        // Hoisted A fragments for this k-step (reused across all 4 n-tiles).
        v16bf ahi[4], alo[4];
#pragma unroll
        for (int mt = 0; mt < 4; ++mt)
          load_a_frag_lds(Ahi, Alo, mt * 16 + lane16, ks, laneh, ahi[mt], alo[mt]);

        const int kstep = (k0 + ks) >> 5;
#pragma unroll
        for (int nt = 0; nt < 4; ++nt) {
          const size_t fidx = ((((size_t)(ntile_base + nt)) * 32 + kstep) * 32 + lane) * 16;
          const v16bf bhi = __builtin_bit_cast(v16bf, *(const v16u*)(Whi + fidx));
          const v16bf blo = __builtin_bit_cast(v16bf, *(const v16u*)(Wlo + fidx));
#pragma unroll
          for (int mt = 0; mt < 4; ++mt)
            C[mt][nt] = wmma3(ahi[mt], alo[mt], bhi, blo, C[mt][nt]);
        }
      }
    }

    // Epilogue: val = v[n]*tanh(C + Uh[b,n]); reduce over n (16 lanes/group).
#pragma unroll
    for (int mt = 0; mt < 4; ++mt) {
#pragma unroll
      for (int nt = 0; nt < 4; ++nt) {
        const int n = n_wave_base + nt * 16 + lane16;
        const float vn = vvec[n];
        const float uh = Uh[(size_t)b * kH + n];
#pragma unroll
        for (int r = 0; r < 8; ++r) {
          float val = vn * tanhf(C[mt][nt][r] + uh);
#pragma unroll
          for (int mask = 1; mask < 16; mask <<= 1)
            val += __shfl_xor(val, mask, 32);
          if (lane16 == 0) {
            const int m = mt * 16 + r + 8 * laneh;
            e_acc[m][wave] += val;  // unique (m,wave) writer; in-order per wave
          }
        }
      }
    }
  }

  __syncthreads();
  if (tid < 64) {
    float s = 0.f;
#pragma unroll
    for (int w = 0; w < 8; ++w) s += e_acc[tid][w];
    e[(size_t)b * kS + s_base + tid] = s;
  }
}

// ---------------------------------------------------------------------------
// Kernel 3: softmax over S per batch (in place). grid = kB, 256 threads.
// ---------------------------------------------------------------------------
__global__ __launch_bounds__(256) void softmax_kernel(float* __restrict__ e) {
  __shared__ float red[256];
  const int b = blockIdx.x;
  const int tid = threadIdx.x;
  const float v0 = e[(size_t)b * kS + tid];
  const float v1 = e[(size_t)b * kS + 256 + tid];
  red[tid] = fmaxf(v0, v1);
  __syncthreads();
  for (int st = 128; st > 0; st >>= 1) {
    if (tid < st) red[tid] = fmaxf(red[tid], red[tid + st]);
    __syncthreads();
  }
  const float mx = red[0];
  __syncthreads();
  const float ex0 = expf(v0 - mx);
  const float ex1 = expf(v1 - mx);
  red[tid] = ex0 + ex1;
  __syncthreads();
  for (int st = 128; st > 0; st >>= 1) {
    if (tid < st) red[tid] += red[tid + st];
    __syncthreads();
  }
  const float inv = 1.f / red[0];
  e[(size_t)b * kS + tid] = ex0 * inv;
  e[(size_t)b * kS + 256 + tid] = ex1 * inv;
}

// ---------------------------------------------------------------------------
// Kernel 4: context[b,h] = sum_s alpha[b,s] * enc[b,s,h]. grid = kB.
// ---------------------------------------------------------------------------
__global__ __launch_bounds__(256) void context_kernel(const float* __restrict__ alpha,
                                                      const float* __restrict__ enc,
                                                      float* __restrict__ ctx) {
  __shared__ float al[kS];
  const int b = blockIdx.x;
  const int tid = threadIdx.x;
  for (int s = tid; s < kS; s += 256) al[s] = alpha[(size_t)b * kS + s];
  __syncthreads();
  float acc[4] = {0.f, 0.f, 0.f, 0.f};
  const float* encb = enc + (size_t)b * kS * kH;
  for (int s = 0; s < kS; ++s) {
    const float a = al[s];
#pragma unroll
    for (int q = 0; q < 4; ++q) acc[q] += a * encb[(size_t)s * kH + q * 256 + tid];
  }
#pragma unroll
  for (int q = 0; q < 4; ++q) ctx[(size_t)b * kH + q * 256 + tid] = acc[q];
}

// ---------------------------------------------------------------------------
// Kernel 5: h[b] = tanh(emb[inp[b]] @ W_ih^T + b_ih + ctx[b] @ W_hh^T + b_hh)
// Writes directly into the h_n section of d_out. grid = kB.
// ---------------------------------------------------------------------------
__global__ __launch_bounds__(256) void rnn_step_kernel(const int* __restrict__ inp,
                                                       const float* __restrict__ emb,
                                                       const float* __restrict__ W_ih,
                                                       const float* __restrict__ W_hh,
                                                       const float* __restrict__ b_ih,
                                                       const float* __restrict__ b_hh,
                                                       const float* __restrict__ ctx,
                                                       float* __restrict__ h_out) {
  __shared__ float xl[kE];
  __shared__ float cl[kH];
  const int b = blockIdx.x;
  const int tid = threadIdx.x;
  const int row = inp[b];
  for (int j = tid; j < kE; j += 256) xl[j] = emb[(size_t)row * kE + j];
  for (int j = tid; j < kH; j += 256) cl[j] = ctx[(size_t)b * kH + j];
  __syncthreads();
#pragma unroll
  for (int q = 0; q < 4; ++q) {
    const int hh = q * 256 + tid;
    float s = b_ih[hh] + b_hh[hh];
    const float* wi = W_ih + (size_t)hh * kE;
    for (int j = 0; j < kE; j += 4) {
      const float4 w = *(const float4*)(wi + j);
      s += w.x * xl[j] + w.y * xl[j + 1] + w.z * xl[j + 2] + w.w * xl[j + 3];
    }
    const float* wh = W_hh + (size_t)hh * kH;
    for (int j = 0; j < kH; j += 4) {
      const float4 w = *(const float4*)(wh + j);
      s += w.x * cl[j] + w.y * cl[j + 1] + w.z * cl[j + 2] + w.w * cl[j + 3];
    }
    h_out[(size_t)b * kH + hh] = tanhf(s);
  }
}

// ---------------------------------------------------------------------------
// Kernel 6: logits(64 x V) = h(64 x 1024) @ out_W^T + out_b, split-bf16 WMMA.
// out_W (206 MB) is streamed once; on-the-fly conversion hides behind HBM.
// grid = ceil(V/512) WGs; each WG covers 512 vocab columns, all 64 batches.
// ---------------------------------------------------------------------------
__global__ __launch_bounds__(256) void logits_kernel(const float* __restrict__ hmat,
                                                     const float* __restrict__ outW,
                                                     const float* __restrict__ outb,
                                                     float* __restrict__ logits) {
  __shared__ unsigned short Ahi[64][128];
  __shared__ unsigned short Alo[64][128];

  const int tid = threadIdx.x;
  const int wave = tid >> 5;
  const int lane = tid & 31;
  const int lane16 = lane & 15;
  const int laneh = lane >> 4;
  const int n_wave_base = blockIdx.x * 512 + wave * 64;

  v8f C[4][4];
#pragma unroll
  for (int mt = 0; mt < 4; ++mt)
#pragma unroll
    for (int nt = 0; nt < 4; ++nt)
      C[mt][nt] = (v8f){0.f, 0.f, 0.f, 0.f, 0.f, 0.f, 0.f, 0.f};

  for (int k0 = 0; k0 < kH; k0 += 128) {
    __syncthreads();
    stage_split64(hmat, k0, Ahi, Alo, tid);
    __syncthreads();

#pragma unroll
    for (int ks = 0; ks < 128; ks += 32) {
      v16bf ahi[4], alo[4];
#pragma unroll
      for (int mt = 0; mt < 4; ++mt)
        load_a_frag_lds(Ahi, Alo, mt * 16 + lane16, ks, laneh, ahi[mt], alo[mt]);

#pragma unroll
      for (int nt = 0; nt < 4; ++nt) {
        int nrow = n_wave_base + nt * 16 + lane16;
        nrow = (nrow < kV) ? nrow : (kV - 1);  // clamp loads; stores guarded
        v16bf bhi, blo;
        load_b_frag_global_f32(outW + (size_t)nrow * kH + (k0 + ks + 16 * laneh), bhi, blo);
#pragma unroll
        for (int mt = 0; mt < 4; ++mt)
          C[mt][nt] = wmma3(ahi[mt], alo[mt], bhi, blo, C[mt][nt]);
      }
    }
  }

#pragma unroll
  for (int mt = 0; mt < 4; ++mt) {
#pragma unroll
    for (int nt = 0; nt < 4; ++nt) {
      const int n = n_wave_base + nt * 16 + lane16;
      if (n < kV) {
        const float bb = outb[n];
#pragma unroll
        for (int r = 0; r < 8; ++r) {
          const int m = mt * 16 + r + 8 * laneh;  // batch index
          logits[(size_t)m * kV + n] = C[mt][nt][r] + bb;
        }
      }
    }
  }
}

// ---------------------------------------------------------------------------
// Host launcher.
// ---------------------------------------------------------------------------
extern "C" void kernel_launch(void* const* d_in, const int* in_sizes, int n_in,
                              void* d_out, int out_size, void* d_ws, size_t ws_size,
                              hipStream_t stream) {
  (void)in_sizes; (void)n_in; (void)out_size; (void)ws_size;

  const int*   inp    = (const int*)  d_in[0];
  const float* hidden = (const float*)d_in[1];
  const float* enc    = (const float*)d_in[2];
  const float* emb    = (const float*)d_in[3];
  const float* W      = (const float*)d_in[4];
  const float* U      = (const float*)d_in[5];
  const float* vvec   = (const float*)d_in[6];
  const float* W_ih   = (const float*)d_in[7];
  const float* W_hh   = (const float*)d_in[8];
  const float* b_ih   = (const float*)d_in[9];
  const float* b_hh   = (const float*)d_in[10];
  const float* out_W  = (const float*)d_in[11];
  const float* out_b  = (const float*)d_in[12];

  float* out    = (float*)d_out;
  float* logits = out;                          // (B,1,V) flat
  float* h_out  = out + (size_t)kB * kV;        // (1,B,H) flat

  // Workspace layout:
  //   [0)                 Uh   : kB*kH floats
  //   [kB*kH)             e    : kB*kS floats (alpha in place)
  //   [kB*kH+kB*kS)       ctx  : kB*kH floats
  //   then                Whi  : kH*kH ushort (2 MB)
  //   then                Wlo  : kH*kH ushort (2 MB)
  float* ws  = (float*)d_ws;
  float* Uh  = ws;
  float* e   = Uh + (size_t)kB * kH;
  float* ctx = e + (size_t)kB * kS;
  unsigned short* Whi = (unsigned short*)(ctx + (size_t)kB * kH);
  unsigned short* Wlo = Whi + (size_t)kH * kH;

  swizzle_w_kernel<<<(kH / 16) * (kH / 32) * 32 / 256, 256, 0, stream>>>(W, Whi, Wlo);
  uh_kernel<<<kB * 4, 256, 0, stream>>>(U, hidden, Uh);
  attn_scores_kernel<<<(kB * kS) / 64, 256, 0, stream>>>(enc, Whi, Wlo, Uh, vvec, e);
  softmax_kernel<<<kB, 256, 0, stream>>>(e);
  context_kernel<<<kB, 256, 0, stream>>>(e, enc, ctx);
  rnn_step_kernel<<<kB, 256, 0, stream>>>(inp, emb, W_ih, W_hh, b_ih, b_hh, ctx, h_out);
  logits_kernel<<<(kV + 511) / 512, 256, 0, stream>>>(h_out, out_W, out_b, logits);
}
